// MultiHeadAttention_22058952032730
// MI455X (gfx1250) — matrix operand order repaired and checked by reading, not compiled, not run on hardware
//
#include <hip/hip_runtime.h>
#include <hip/hip_bf16.h>

// ---------------------------------------------------------------------------
// Types / helpers for CDNA5 WMMA (gfx1250, wave32)
// ---------------------------------------------------------------------------
typedef __attribute__((ext_vector_type(16))) __bf16 v16bf;
typedef __attribute__((ext_vector_type(8)))  float  v8f;

union Frag16 {            // 16 bf16 = 32 bytes = 2 x uint4
    v16bf v;
    uint4 q[2];
};

__device__ __forceinline__ unsigned short f2bf(float f) {
    // native f32 -> bf16 (RNE fptrunc -> v_cvt)
    return __builtin_bit_cast(unsigned short, (__bf16)f);
}

__device__ __forceinline__ v8f wmma_bf16(v16bf a, v16bf b, v8f c) {
    // D = A(16x32 bf16) * B(32x16 bf16) + C(16x16 f32)
    return __builtin_amdgcn_wmma_f32_16x16x32_bf16(
        /*neg_a=*/false, a, /*neg_b=*/false, b,
        /*c_mod=*/(short)0, c, /*reuse_a=*/false, /*reuse_b=*/false);
}

#define NEG_INF (-4294967295.0f)   // -2^32 + 1, matches reference
#define LOG2E   1.4426950408889634f

// Problem constants
#define Bc 2
#define Sc 2048
#define Dc 1024
#define Hc 16
#define DKc 64
#define DVc 64

// ---------------------------------------------------------------------------
// f32 -> bf16 conversion (x)
// ---------------------------------------------------------------------------
__global__ void cvt_f32_bf16(const float* __restrict__ x,
                             unsigned short* __restrict__ y, int n) {
    int i = (blockIdx.x * blockDim.x + threadIdx.x) * 4;
    if (i + 3 < n) {
        float4 f = *(const float4*)(x + i);
        y[i + 0] = f2bf(f.x);
        y[i + 1] = f2bf(f.y);
        y[i + 2] = f2bf(f.z);
        y[i + 3] = f2bf(f.w);
    } else {
        for (; i < n; ++i) y[i] = f2bf(x[i]);
    }
}

// ---------------------------------------------------------------------------
// f32 [K][N] -> bf16 transposed [N][K], optionally scaled (weights; K=N=1024)
// scale = LOG2E for Wq folds the softmax exp->exp2 conversion into Q.
// ---------------------------------------------------------------------------
__global__ __launch_bounds__(256) void transpose_cvt(
    const float* __restrict__ W, unsigned short* __restrict__ Wt, int K, int N,
    float scale) {
    __shared__ float tile[32][33];
    const int tx = threadIdx.x, ty = threadIdx.y;   // block (32, 8)
    const int x = blockIdx.x * 32 + tx;             // N index
    const int y = blockIdx.y * 32 + ty;             // K index
#pragma unroll
    for (int j = 0; j < 4; ++j)
        tile[ty + j * 8][tx] = W[(size_t)(y + j * 8) * N + x];
    __syncthreads();
    const int xo = blockIdx.y * 32 + tx;            // K index (now contiguous)
    const int yo = blockIdx.x * 32 + ty;            // N index (row of Wt)
#pragma unroll
    for (int j = 0; j < 4; ++j)
        Wt[(size_t)(yo + j * 8) * K + xo] = f2bf(tile[tx][ty + j * 8] * scale);
}

// ---------------------------------------------------------------------------
// GEMM: C[M,N] = A[M,K](bf16) * Wt[N,K]^T(bf16) + bias*bias_scale
// 128 threads = 4 waves; block tile 128x64; wave tile 32x64; K step 32.
// Per K-step / wave: 8 WMMAs vs 12 b128 loads.
// MODE 0: bf16 row-major out   MODE 1: bf16 V-transposed out [b][h][dv][s]
// MODE 2: f32 row-major out
// ---------------------------------------------------------------------------
template <int MODE>
__global__ __launch_bounds__(128) void gemm_wmma(
    const unsigned short* __restrict__ A, const unsigned short* __restrict__ Bt,
    const float* __restrict__ bias, void* __restrict__ outp,
    int M, int N, int K, float bias_scale) {
    const int lane = threadIdx.x & 31;
    const int wave = threadIdx.x >> 5;
    const int gm   = blockIdx.y * 128 + wave * 32;
    const int gn   = blockIdx.x * 64;
    const int l15  = lane & 15;
    const int kb8  = (lane >> 4) * 8;    // A-layout K sub-offset
    const int kb16 = (lane >> 4) * 16;   // unused (B fragments also use kb8)
    const size_t arow0 = (size_t)(gm + l15) * K;
    const size_t arow1 = (size_t)(gm + 16 + l15) * K;

    v8f acc[2][4] = {};

    for (int kt = 0; kt < K; kt += 32) {
        // A fragments (two 16x32 row-strips)
        const unsigned short* ap0 = A + arow0 + kt + kb8;
        const unsigned short* ap1 = A + arow1 + kt + kb8;
        Frag16 af0, af1;
        af0.q[0] = *(const uint4*)(ap0);
        af0.q[1] = *(const uint4*)(ap0 + 16);
        af1.q[0] = *(const uint4*)(ap1);
        af1.q[1] = *(const uint4*)(ap1 + 16);
#pragma unroll
        for (int n = 0; n < 4; ++n) {
            // B fragment (32x16): col = gn + n*16 + l15, K in A (table) order
            const unsigned short* bp =
                Bt + (size_t)(gn + n * 16 + l15) * K + kt + kb8;
            Frag16 bf;
            bf.q[0] = ((const uint4*)bp)[0];
            bf.q[1] = ((const uint4*)(bp + 16))[0];
            acc[0][n] = wmma_bf16(af0.v, bf.v, acc[0][n]);
            acc[1][n] = wmma_bf16(af1.v, bf.v, acc[1][n]);
        }
    }

    // Epilogue: C layout -> row = (lane>>4)*8 + i, col = l15 (+16n)
#pragma unroll
    for (int r = 0; r < 2; ++r) {
#pragma unroll
        for (int n = 0; n < 4; ++n) {
            const int colg = gn + n * 16 + l15;
            const float bv = bias[colg] * bias_scale;
#pragma unroll
            for (int i = 0; i < 8; ++i) {
                const int rowg = gm + r * 16 + (lane >> 4) * 8 + i;
                const float val = acc[r][n][i] + bv;
                if (MODE == 0) {
                    ((unsigned short*)outp)[(size_t)rowg * N + colg] = f2bf(val);
                } else if (MODE == 1) {
                    const int bb = rowg >> 11, s = rowg & (Sc - 1);
                    const int h = colg >> 6, dv = colg & (DVc - 1);
                    ((unsigned short*)outp)
                        [((size_t)((bb * Hc + h) * DVc + dv) << 11) + s] = f2bf(val);
                } else {
                    ((float*)outp)[(size_t)rowg * N + colg] = val;
                }
            }
        }
    }
}

// ---------------------------------------------------------------------------
// Flash attention (causal, no 1/sqrt(dk) scale), bf16 WMMA, online softmax.
// Scores arrive pre-scaled by log2e (folded into Wq/bq), so softmax uses
// exp2 directly (v_exp_f32 is exp2 on AMD) -- mathematically identical.
// grid (S/64, H, B), 128 threads = 4 waves; each wave owns 16 query rows.
// 64 keys per iteration: 8 score WMMAs + 8 P*V WMMAs between barriers.
// Causal split: tiles kt < qb fully unmasked; only kt == qb applies the mask.
// Q,K: bf16 [B*S][H*64] row-major.  Vt: bf16 [B][H][DV][S].  ctx: bf16 [B*S][H*64].
// ---------------------------------------------------------------------------
__global__ __launch_bounds__(128) void flash_attn(
    const unsigned short* __restrict__ Q, const unsigned short* __restrict__ Km,
    const unsigned short* __restrict__ Vt, unsigned short* __restrict__ ctx) {
    __shared__ unsigned short Plds[4][16][72];   // per-wave 16x64 P tile, padded

    const int lane = threadIdx.x & 31;
    const int wave = threadIdx.x >> 5;
    const int b = blockIdx.z, h = blockIdx.y;
    const int qb = blockIdx.x * 64;
    const int q0 = qb + wave * 16;
    const int l15 = lane & 15, hi = lane >> 4;
    const int kb8 = hi * 8, kb16 = hi * 16;
    const int HD = Hc * DKc;                     // 1024

    // Q fragments: dk 0..31 and 32..63 (A layout)
    const unsigned short* qrow = Q + (size_t)(b * Sc + q0 + l15) * HD + h * DKc;
    Frag16 qa0, qa1;
    qa0.q[0] = *(const uint4*)(qrow + kb8);
    qa0.q[1] = *(const uint4*)(qrow + 16 + kb8);
    qa1.q[0] = *(const uint4*)(qrow + 32 + kb8);
    qa1.q[1] = *(const uint4*)(qrow + 48 + kb8);

    float m[8], l[8];
#pragma unroll
    for (int i = 0; i < 8; ++i) { m[i] = NEG_INF; l[i] = 0.0f; }
    v8f o[4] = {};

    auto process_tile = [&](int kt, bool masked) __attribute__((always_inline)) {
        // ---- scores: S = Q * K^T, four 16x16 key-subtiles
        v8f s[4];
#pragma unroll
        for (int t = 0; t < 4; ++t) {
            const unsigned short* kr =
                Km + (size_t)(b * Sc + kt + t * 16 + l15) * HD + h * DKc + kb8;
            Frag16 kf0, kf1;
            kf0.q[0] = ((const uint4*)kr)[0];
            kf0.q[1] = ((const uint4*)(kr + 16))[0];
            kf1.q[0] = ((const uint4*)(kr + 32))[0];
            kf1.q[1] = ((const uint4*)(kr + 48))[0];
            v8f z = {};
            z = wmma_bf16(qa0.v, kf0.v, z);
            s[t] = wmma_bf16(qa1.v, kf1.v, z);
        }

        // ---- (masked) online softmax in exp2 domain; rows in 16-lane halves
        float corr[8];
#pragma unroll
        for (int i = 0; i < 8; ++i) {
            float v[4];
            if (masked) {
                const int qg = q0 + hi * 8 + i;
#pragma unroll
                for (int t = 0; t < 4; ++t)
                    v[t] = ((kt + t * 16 + l15) <= qg) ? s[t][i] : NEG_INF;
            } else {
#pragma unroll
                for (int t = 0; t < 4; ++t) v[t] = s[t][i];
            }
            float mx = fmaxf(fmaxf(v[0], v[1]), fmaxf(v[2], v[3]));
            mx = fmaxf(mx, __shfl_xor(mx, 1, 32));
            mx = fmaxf(mx, __shfl_xor(mx, 2, 32));
            mx = fmaxf(mx, __shfl_xor(mx, 4, 32));
            mx = fmaxf(mx, __shfl_xor(mx, 8, 32));
            const float mn = fmaxf(m[i], mx);
            float p[4];
            float rs = 0.0f;
#pragma unroll
            for (int t = 0; t < 4; ++t) { p[t] = exp2f(v[t] - mn); rs += p[t]; }
            rs += __shfl_xor(rs, 1, 32);
            rs += __shfl_xor(rs, 2, 32);
            rs += __shfl_xor(rs, 4, 32);
            rs += __shfl_xor(rs, 8, 32);
            const float c = exp2f(m[i] - mn);
            l[i] = l[i] * c + rs;
            m[i] = mn;
            corr[i] = c;
            const int prow = hi * 8 + i;                 // C-layout row
#pragma unroll
            for (int t = 0; t < 4; ++t)
                Plds[wave][prow][t * 16 + l15] = f2bf(p[t]);
        }

        // rescale running output by correction factor
#pragma unroll
        for (int n = 0; n < 4; ++n)
#pragma unroll
            for (int i = 0; i < 8; ++i) o[n][i] *= corr[i];

        __syncthreads();   // P write -> P read ordering (uniform trip count)

        // ---- P fragments in A layout (16 queries x 64 keys) from LDS
        Frag16 pa0, pa1;
        const unsigned short* pr = &Plds[wave][l15][0];
        pa0.q[0] = *(const uint4*)(pr + kb8);
        pa0.q[1] = *(const uint4*)(pr + 16 + kb8);
        pa1.q[0] = *(const uint4*)(pr + 32 + kb8);
        pa1.q[1] = *(const uint4*)(pr + 48 + kb8);

        // ---- O += P * V : 4 dv-subtiles x 2 key-halves
#pragma unroll
        for (int n = 0; n < 4; ++n) {
            const unsigned short* vp =
                Vt + ((size_t)((b * Hc + h) * DVc + n * 16 + l15) << 11) + kt + kb8;
            Frag16 vf0, vf1;
            vf0.q[0] = ((const uint4*)vp)[0];
            vf0.q[1] = ((const uint4*)(vp + 16))[0];
            vf1.q[0] = ((const uint4*)(vp + 32))[0];
            vf1.q[1] = ((const uint4*)(vp + 48))[0];
            o[n] = wmma_bf16(pa0.v, vf0.v, o[n]);
            o[n] = wmma_bf16(pa1.v, vf1.v, o[n]);
        }
        __syncthreads();
    };

    // Unmasked main loop (strictly below the diagonal block), uniform per block
    for (int kt = 0; kt < qb; kt += 64) process_tile(kt, false);
    // Diagonal tile: the only one where the causal mask is active
    process_tile(qb, true);

    // ---- epilogue: ctx = O / l
#pragma unroll
    for (int i = 0; i < 8; ++i) {
        const int rowg = b * Sc + q0 + hi * 8 + i;
        const float inv = 1.0f / l[i];
#pragma unroll
        for (int n = 0; n < 4; ++n) {
            ctx[(size_t)rowg * HD + h * DVc + n * 16 + l15] = f2bf(o[n][i] * inv);
        }
    }
}

// ---------------------------------------------------------------------------
// Host-side launcher
// ---------------------------------------------------------------------------
extern "C" void kernel_launch(void* const* d_in, const int* in_sizes, int n_in,
                              void* d_out, int out_size, void* d_ws, size_t ws_size,
                              hipStream_t stream) {
    (void)in_sizes; (void)n_in; (void)out_size; (void)ws_size;
    const float* x  = (const float*)d_in[0];
    const float* Wq = (const float*)d_in[1];
    const float* bq = (const float*)d_in[2];
    const float* Wk = (const float*)d_in[3];
    const float* bk = (const float*)d_in[4];
    const float* Wv = (const float*)d_in[5];
    const float* bv = (const float*)d_in[6];
    const float* Wo = (const float*)d_in[7];
    const float* bo = (const float*)d_in[8];
    float* out = (float*)d_out;

    const size_t MS = (size_t)Bc * Sc;     // 4096
    const size_t DD = Dc;                  // 1024
    char* ws = (char*)d_ws;
    unsigned short* xbf  = (unsigned short*)ws;  ws += MS * DD * 2;
    unsigned short* wqT  = (unsigned short*)ws;  ws += DD * DD * 2;
    unsigned short* wkT  = (unsigned short*)ws;  ws += DD * DD * 2;
    unsigned short* wvT  = (unsigned short*)ws;  ws += DD * DD * 2;
    unsigned short* woT  = (unsigned short*)ws;  ws += DD * DD * 2;
    unsigned short* Qb   = (unsigned short*)ws;  ws += MS * DD * 2;
    unsigned short* Kb   = (unsigned short*)ws;  ws += MS * DD * 2;
    unsigned short* Vtb  = (unsigned short*)ws;  ws += MS * DD * 2;
    unsigned short* ctxb = (unsigned short*)ws;  ws += MS * DD * 2;

    // 1) precision conversion + weight transposes (Wq pre-scaled by log2e)
    cvt_f32_bf16<<<dim3((unsigned)(MS * DD / (256 * 4))), dim3(256), 0, stream>>>(
        x, xbf, (int)(MS * DD));
    transpose_cvt<<<dim3(32, 32), dim3(32, 8), 0, stream>>>(Wq, wqT, Dc, Dc, LOG2E);
    transpose_cvt<<<dim3(32, 32), dim3(32, 8), 0, stream>>>(Wk, wkT, Dc, Dc, 1.0f);
    transpose_cvt<<<dim3(32, 32), dim3(32, 8), 0, stream>>>(Wv, wvT, Dc, Dc, 1.0f);
    transpose_cvt<<<dim3(32, 32), dim3(32, 8), 0, stream>>>(Wo, woT, Dc, Dc, 1.0f);

    // 2) QKV projections (V written transposed per head for flash B-fragments)
    dim3 ggrid(Dc / 64, (unsigned)(MS / 128));
    gemm_wmma<0><<<ggrid, dim3(128), 0, stream>>>(xbf, wqT, bq, Qb,
                                                  (int)MS, Dc, Dc, LOG2E);
    gemm_wmma<0><<<ggrid, dim3(128), 0, stream>>>(xbf, wkT, bk, Kb,
                                                  (int)MS, Dc, Dc, 1.0f);
    gemm_wmma<1><<<ggrid, dim3(128), 0, stream>>>(xbf, wvT, bv, Vtb,
                                                  (int)MS, Dc, Dc, 1.0f);

    // 3) causal flash attention (exp2-domain softmax)
    flash_attn<<<dim3(Sc / 64, Hc, Bc), dim3(128), 0, stream>>>(Qb, Kb, Vtb, ctxb);

    // 4) output projection -> f32
    gemm_wmma<2><<<ggrid, dim3(128), 0, stream>>>(ctxb, woT, bo, out,
                                                  (int)MS, Dc, Dc, 1.0f);
}
